// Attention3D_25769803835
// MI455X (gfx1250) — matrix operand order repaired and checked by reading, not compiled, not run on hardware
//
#include <hip/hip_runtime.h>
#include <hip/hip_bf16.h>
#include <math.h>

// ---------------- CDNA5 WMMA types ----------------
typedef __attribute__((ext_vector_type(16))) __bf16 v16bf;
typedef __attribute__((ext_vector_type(2)))  __bf16 v2bf;
typedef __attribute__((ext_vector_type(8)))  float  v8f;
typedef __attribute__((ext_vector_type(4)))  unsigned int u32x4;

union BF16Frag {
    v16bf v;
    unsigned short h[16];
    unsigned int   u[8];
};

static __device__ __forceinline__ unsigned short f2bf(float f) {
    unsigned int u = __float_as_uint(f);
    u += 0x7FFFu + ((u >> 16) & 1u);   // round-to-nearest-even
    return (unsigned short)(u >> 16);
}

static __device__ __forceinline__ unsigned int f2bf_pk(float lo, float hi) {
#if __has_builtin(__builtin_amdgcn_cvt_pk_bf16_f32)
    union { v2bf v; unsigned int u; } x;
    x.v = __builtin_amdgcn_cvt_pk_bf16_f32(lo, hi);
    return x.u;
#else
    return (unsigned int)f2bf(lo) | ((unsigned int)f2bf(hi) << 16);
#endif
}

static __device__ __forceinline__ v8f wmma_bf16(const BF16Frag& a, const BF16Frag& b, v8f c) {
    return __builtin_amdgcn_wmma_f32_16x16x32_bf16(false, a.v, false, b.v, (short)0, c, false, false);
}

// async global->LDS copy helpers (CDNA5 ASYNCcnt path, inline asm = toolchain-portable)
static __device__ __forceinline__ void async_ld_b128(unsigned lds_off, const void* gaddr) {
    asm volatile("global_load_async_to_lds_b128 %0, %1, off"
                 :: "v"(lds_off), "v"((unsigned long long)gaddr) : "memory");
}
static __device__ __forceinline__ void async_ld_b32(unsigned lds_off, const void* gaddr, int ioff) {
    if (ioff == 0)
        asm volatile("global_load_async_to_lds_b32 %0, %1, off"
                     :: "v"(lds_off), "v"((unsigned long long)gaddr) : "memory");
    else if (ioff == 4)
        asm volatile("global_load_async_to_lds_b32 %0, %1, off offset:4"
                     :: "v"(lds_off), "v"((unsigned long long)gaddr) : "memory");
    else if (ioff == 8)
        asm volatile("global_load_async_to_lds_b32 %0, %1, off offset:8"
                     :: "v"(lds_off), "v"((unsigned long long)gaddr) : "memory");
    else
        asm volatile("global_load_async_to_lds_b32 %0, %1, off offset:12"
                     :: "v"(lds_off), "v"((unsigned long long)gaddr) : "memory");
}
static __device__ __forceinline__ void wait_asynccnt0() {
    asm volatile("s_wait_asynccnt 0" ::: "memory");
}

// ---------------- problem constants ----------------
#define BB    2
#define CC    256
#define NN    4096
#define HEADS 8
#define DH    64
#define HID   512
#define MQKV  1536   // 3*HID

// LDS row strides (dwords) padded for 64-bank conflict-free b128 access
#define KT_STRIDE 36   // ktile: 32 rows
#define VT_STRIDE 12   // vtile: per (t,lane) group of 8 dwords
#define PT_STRIDE 40   // pt: row stride in u16 (= 20 dwords)

// ---------------- kernel 1: fused norm + transpose-pack of x ----------------
__global__ void norm_pack_x_kernel(const float* __restrict__ x, float* __restrict__ s,
                                   unsigned int* __restrict__ xt32) {
    int idx = blockIdx.x * blockDim.x + threadIdx.x;   // [0, B*N)
    int b = idx >> 12;
    int n = idx & (NN - 1);
    const float* xp = x + (size_t)b * CC * NN + n;
    unsigned int* xr = xt32 + (size_t)idx * (CC / 2);
    float acc = 0.f;
    for (int c8 = 0; c8 < CC / 8; ++c8) {
        u32x4 pk;
        #pragma unroll
        for (int q = 0; q < 4; ++q) {
            float v0 = xp[(size_t)(c8 * 8 + 2 * q) * NN];
            float v1 = xp[(size_t)(c8 * 8 + 2 * q + 1) * NN];
            acc = fmaf(v0, v0, acc);
            acc = fmaf(v1, v1, acc);
            pk[q] = f2bf_pk(v0, v1);
        }
        *(u32x4*)(xr + c8 * 4) = pk;
    }
    float nrm = fmaxf(sqrtf(acc), 1e-12f);
    s[idx] = 16.0f / nrm;   // sqrt(256) = 16
}

// ---------------- pack kernels: one-shot weight conversion ----------------
__global__ void pack_wqkv_kernel(const float* __restrict__ wqkv, const float* __restrict__ g,
                                 unsigned int* __restrict__ wg32) {
    int i = blockIdx.x * 256 + threadIdx.x;   // [0, 1536*128)
    int o = i >> 7, c = (i & 127) * 2;
    wg32[i] = f2bf_pk(wqkv[(size_t)o * CC + c] * g[c], wqkv[(size_t)o * CC + c + 1] * g[c + 1]);
}

__global__ void pack_wout_kernel(const float* __restrict__ wout, unsigned int* __restrict__ wo32) {
    int i = blockIdx.x * 256 + threadIdx.x;   // [0, 256*256)
    wo32[i] = f2bf_pk(wout[2 * i], wout[2 * i + 1]);
}

// ---------------- kernel 2: QKV GEMM (WMMA bf16) ----------------
__global__ void qkv_kernel(const unsigned int* __restrict__ xt32,
                           const unsigned int* __restrict__ wg32,
                           const float* __restrict__ s,
                           unsigned short* __restrict__ qbuf,
                           unsigned short* __restrict__ kbuf,
                           unsigned short* __restrict__ vpbuf) {
    const int lane = threadIdx.x & 31;
    const int wave = threadIdx.x >> 5;
    const int b    = blockIdx.z;
    const int m0   = blockIdx.y * 128 + wave * 16;
    const int n0   = blockIdx.x * 64;
    const int half = lane >> 4;
    const int ncol = lane & 15;
    const int mrow = m0 + ncol;

    v8f acc[4] = {};
    for (int kc = 0; kc < CC / 32; ++kc) {
        const int k0 = kc * 32;
        BF16Frag a;
        {
            const unsigned int* ap = wg32 + (size_t)mrow * 128 + (k0 >> 1) + half * 8;
            *(u32x4*)&a.u[0] = *(const u32x4*)ap;
            *(u32x4*)&a.u[4] = *(const u32x4*)(ap + 4);
        }
        #pragma unroll
        for (int nt = 0; nt < 4; ++nt) {
            BF16Frag bb;
            int n = n0 + nt * 16 + ncol;
            const unsigned int* bp =
                xt32 + ((size_t)b * NN + n) * (CC / 2) + (k0 >> 1) + half * 8;
            *(u32x4*)&bb.u[0] = *(const u32x4*)bp;
            *(u32x4*)&bb.u[4] = *(const u32x4*)(bp + 4);
            acc[nt] = wmma_bf16(a, bb, acc[nt]);
        }
    }

    const int mat = m0 >> 9;   // wave-uniform: 0=Q 1=K 2=V
    #pragma unroll
    for (int nt = 0; nt < 4; ++nt) {
        int n = n0 + nt * 16 + ncol;
        float sc = s[b * NN + n];
        if (mat < 2) {
            unsigned short* dst = (mat == 0) ? qbuf : kbuf;
            #pragma unroll
            for (int v = 0; v < 8; v += 2) {
                int o = m0 + v + 8 * half;
                int head = (o >> 6) & 7, d = o & 63;   // d even; pair inside one head
                *(unsigned int*)(dst + (((size_t)(b * HEADS + head) * NN + n) * DH + d)) =
                    f2bf_pk(acc[nt][v] * sc, acc[nt][v + 1] * sc);
            }
        } else {
            #pragma unroll
            for (int v = 0; v < 8; ++v) {
                int o = m0 + v + 8 * half;
                int head = (o >> 6) & 7, d = o & 63;
                vpbuf[(((size_t)(b * HEADS + head) * (NN / 2) + (n >> 1)) * DH + d) * 2 + (n & 1)] =
                    f2bf(acc[nt][v] * sc);
            }
        }
    }
}

// ---------------- kernel 3: flash attention (WMMA bf16, async-LDS staged K/V) ----------------
__global__ void attn_kernel(const unsigned short* __restrict__ qbuf,
                            const unsigned short* __restrict__ kbuf,
                            const unsigned short* __restrict__ vpbuf,
                            unsigned short* __restrict__ obuf) {
    __shared__ unsigned int   ktile[32 * KT_STRIDE];       // [j(32)][dpair(32)] padded
    __shared__ unsigned int   vtile[4 * 32 * VT_STRIDE];   // frag-ready [(t*32+lane)][j(8)] padded
    __shared__ unsigned short pt[8][16 * PT_STRIDE];       // per-wave P tile, padded rows

    const int tid  = threadIdx.x;
    const int lane = tid & 31;
    const int wave = tid >> 5;
    const int b    = blockIdx.z;
    const int head = blockIdx.y;
    const int q0   = blockIdx.x * 128 + wave * 16;
    const int half = lane >> 4;
    const int ncol = lane & 15;

    const size_t hb32 = (size_t)(b * HEADS + head) * (NN * DH / 2);   // u32 units
    const unsigned int* Qh32 = (const unsigned int*)qbuf  + hb32;
    const unsigned int* Kh32 = (const unsigned int*)kbuf  + hb32;
    const unsigned int* Vh32 = (const unsigned int*)vpbuf + hb32;

    // cooperative-copy decompositions (per thread)
    const int kj  = tid >> 3, kd4 = (tid & 7) * 4;         // K: row, dpair*4
    const int vp_ = tid >> 4, vd  = (tid & 15) * 4;        // V: pair-row, d*4
    const int vt_ = vd >> 4;
    const int vbase = ((vt_ * 32) + ((vp_ >> 3) * 16) + (vd & 15)) * VT_STRIDE + (vp_ & 7);

    // per-lane LDS byte offsets for async copies
    const unsigned kdst  = (unsigned)(size_t)&ktile[kj * KT_STRIDE + kd4];
    const unsigned vdst0 = (unsigned)(size_t)&vtile[vbase];

    // Q A-fragments (d=0..31, 32..63), 2x b128 each, kept resident
    BF16Frag qa[2];
    {
        int m = q0 + ncol;
        #pragma unroll
        for (int t = 0; t < 2; ++t) {
            const unsigned int* ap = Qh32 + (size_t)m * 32 + t * 16 + half * 8;
            *(u32x4*)&qa[t].u[0] = *(const u32x4*)ap;
            *(u32x4*)&qa[t].u[4] = *(const u32x4*)(ap + 4);
        }
    }

    v8f ofr[4] = {};
    float mrow[8], lrow[8];
    #pragma unroll
    for (int v = 0; v < 8; ++v) { mrow[v] = -3.0e38f; lrow[v] = 0.f; }

    const float SCL = 0.125f * 1.44269504089f;   // dh^-0.5 * log2(e); softmax in log2 domain

    for (int j0 = 0; j0 < NN; j0 += 32) {
        // K tile: async b128 copy (contiguous global -> padded-row LDS)
        async_ld_b128(kdst, Kh32 + (size_t)j0 * 32 + tid * 4);
        // V tile: async b32 scatter to fragment-ready order.
        // INST_OFFSET adds to BOTH global and LDS addresses, so bias the LDS base
        // by q*stride*4 - 4q and use offset:4q to walk src dwords / dst rows together.
        {
            const void* vsrc = Vh32 + (size_t)((j0 >> 1) + vp_) * 64 + vd;
            async_ld_b32(vdst0, vsrc, 0);
            async_ld_b32(vdst0 + VT_STRIDE * 4 - 4, vsrc, 4);
            async_ld_b32(vdst0 + 2 * VT_STRIDE * 4 - 8, vsrc, 8);
            async_ld_b32(vdst0 + 3 * VT_STRIDE * 4 - 12, vsrc, 12);
        }
        if (j0 + 32 < NN) {
            __builtin_prefetch(Kh32 + (size_t)(j0 + 32) * 32 + tid * 4, 0, 0);
            __builtin_prefetch(Vh32 + (size_t)((j0 + 32) >> 1) * 64 + tid * 4, 0, 0);
        }
        wait_asynccnt0();
        __syncthreads();

        // S = (Q*scale) K^T : 2 key sub-tiles of 16, K-frags 2x ds_load_b128
        v8f s0 = {}, s1 = {};
        #pragma unroll
        for (int t = 0; t < 2; ++t) {
            BF16Frag kb;
            int dpb = t * 16 + half * 8;
            *(u32x4*)&kb.u[0] = *(const u32x4*)&ktile[ncol * KT_STRIDE + dpb];
            *(u32x4*)&kb.u[4] = *(const u32x4*)&ktile[ncol * KT_STRIDE + dpb + 4];
            s0 = wmma_bf16(qa[t], kb, s0);
            *(u32x4*)&kb.u[0] = *(const u32x4*)&ktile[(16 + ncol) * KT_STRIDE + dpb];
            *(u32x4*)&kb.u[4] = *(const u32x4*)&ktile[(16 + ncol) * KT_STRIDE + dpb + 4];
            s1 = wmma_bf16(qa[t], kb, s1);
        }

        // online softmax in log2 domain (rows = v + 8*half; 16 cols per lane-group)
        #pragma unroll
        for (int v = 0; v < 8; ++v) {
            float a = s0[v] * SCL;
            float c = s1[v] * SCL;
            float mv = fmaxf(a, c);
            mv = fmaxf(mv, __shfl_xor(mv, 1, 32));
            mv = fmaxf(mv, __shfl_xor(mv, 2, 32));
            mv = fmaxf(mv, __shfl_xor(mv, 4, 32));
            mv = fmaxf(mv, __shfl_xor(mv, 8, 32));
            float mnew = fmaxf(mrow[v], mv);
            float corr = exp2f(mrow[v] - mnew);
            float p0 = exp2f(a - mnew);
            float p1 = exp2f(c - mnew);
            s0[v] = p0;
            s1[v] = p1;
            float rs = p0 + p1;
            rs += __shfl_xor(rs, 1, 32);
            rs += __shfl_xor(rs, 2, 32);
            rs += __shfl_xor(rs, 4, 32);
            rs += __shfl_xor(rs, 8, 32);
            lrow[v] = lrow[v] * corr + rs;
            mrow[v] = mnew;
            #pragma unroll
            for (int t = 0; t < 4; ++t) ofr[t][v] *= corr;
        }

        // C-layout -> A-layout via per-wave LDS tile (in-order DS, wave-private)
        #pragma unroll
        for (int v = 0; v < 8; ++v) {
            int m = v + 8 * half;
            pt[wave][m * PT_STRIDE + ncol]      = f2bf(s0[v]);
            pt[wave][m * PT_STRIDE + 16 + ncol] = f2bf(s1[v]);
        }
        BF16Frag pa;
        {
            const unsigned short* pp = &pt[wave][ncol * PT_STRIDE + half * 16];
            *(u32x4*)&pa.u[0] = *(const u32x4*)pp;
            *(u32x4*)&pa.u[4] = *(const u32x4*)(pp + 8);
        }

        // O += P @ V : V-frags 2x ds_load_b128 from fragment-ready tile
        #pragma unroll
        for (int t = 0; t < 4; ++t) {
            BF16Frag vb;
            const unsigned int* vp = &vtile[(t * 32 + lane) * VT_STRIDE];
            *(u32x4*)&vb.u[0] = *(const u32x4*)vp;
            *(u32x4*)&vb.u[4] = *(const u32x4*)(vp + 4);
            ofr[t] = wmma_bf16(pa, vb, ofr[t]);
        }
        __syncthreads();
    }

    // normalize and store bf16 [b][head][n][d]
    #pragma unroll
    for (int v = 0; v < 8; ++v) {
        float inv = 1.0f / lrow[v];
        int m = q0 + v + 8 * half;
        #pragma unroll
        for (int t = 0; t < 4; ++t) {
            obuf[((size_t)(b * HEADS + head) * NN + m) * DH + t * 16 + ncol] =
                f2bf(ofr[t][v] * inv);
        }
    }
}

// ---------------- kernel 4: output projection (WMMA bf16) + bias ----------------
__global__ void out_kernel(const unsigned int* __restrict__ wo32, const float* __restrict__ bout,
                           const unsigned int* __restrict__ ob32, float* __restrict__ y) {
    const int lane = threadIdx.x & 31;
    const int wave = threadIdx.x >> 5;
    const int b    = blockIdx.z;
    const int m0   = blockIdx.y * 128 + wave * 16;
    const int n0   = blockIdx.x * 64;
    const int half = lane >> 4;
    const int ncol = lane & 15;
    const int mrow = m0 + ncol;

    v8f acc[4] = {};
    for (int kc = 0; kc < HID / 32; ++kc) {
        const int k0 = kc * 32;
        BF16Frag a;
        {
            const unsigned int* ap = wo32 + (size_t)mrow * 256 + (k0 >> 1) + half * 8;
            *(u32x4*)&a.u[0] = *(const u32x4*)ap;
            *(u32x4*)&a.u[4] = *(const u32x4*)(ap + 4);
        }
        const int kk   = k0 + half * 16;    // 16-chunk stays inside one head
        const int head = kk >> 6;
        const int dh2  = (kk & 63) >> 1;
        #pragma unroll
        for (int nt = 0; nt < 4; ++nt) {
            BF16Frag bb;
            int n = n0 + nt * 16 + ncol;
            const unsigned int* bp = ob32 + ((size_t)(b * HEADS + head) * NN + n) * 32 + dh2;
            *(u32x4*)&bb.u[0] = *(const u32x4*)bp;
            *(u32x4*)&bb.u[4] = *(const u32x4*)(bp + 4);
            acc[nt] = wmma_bf16(a, bb, acc[nt]);
        }
    }
    #pragma unroll
    for (int nt = 0; nt < 4; ++nt) {
        int n = n0 + nt * 16 + ncol;
        #pragma unroll
        for (int v = 0; v < 8; ++v) {
            int o = m0 + v + 8 * half;
            y[((size_t)b * CC + o) * NN + n] = acc[nt][v] + bout[o];
        }
    }
}

// ---------------- launcher ----------------
extern "C" void kernel_launch(void* const* d_in, const int* in_sizes, int n_in,
                              void* d_out, int out_size, void* d_ws, size_t ws_size,
                              hipStream_t stream) {
    const float* x    = (const float*)d_in[0];
    const float* g    = (const float*)d_in[1];
    const float* wqkv = (const float*)d_in[2];
    const float* wout = (const float*)d_in[3];
    const float* bout = (const float*)d_in[4];
    float* y = (float*)d_out;

    // workspace layout (bytes)
    char* ws = (char*)d_ws;
    const size_t OFF_WG   = 32768;                                     // s: 2*4096 f32
    const size_t OFF_WOUT = OFF_WG   + (size_t)MQKV * CC * 2;          // 786432
    const size_t OFF_XT   = OFF_WOUT + (size_t)CC * HID * 2;           // 262144
    const size_t OFF_Q    = OFF_XT   + (size_t)BB * CC * NN * 2;       // 4 MB
    const size_t SZ_HNB   = (size_t)BB * HEADS * NN * DH * 2;          // 8 MB each
    const size_t OFF_K    = OFF_Q + SZ_HNB;
    const size_t OFF_V    = OFF_K + SZ_HNB;
    const size_t OFF_O    = OFF_V + SZ_HNB;

    float*          s     = (float*)ws;
    unsigned int*   wg32  = (unsigned int*)(ws + OFF_WG);
    unsigned int*   wo32  = (unsigned int*)(ws + OFF_WOUT);
    unsigned int*   xt32  = (unsigned int*)(ws + OFF_XT);
    unsigned short* qbuf  = (unsigned short*)(ws + OFF_Q);
    unsigned short* kbuf  = (unsigned short*)(ws + OFF_K);
    unsigned short* vpbuf = (unsigned short*)(ws + OFF_V);
    unsigned short* obuf  = (unsigned short*)(ws + OFF_O);

    norm_pack_x_kernel<<<dim3((BB * NN) / 256), dim3(256), 0, stream>>>(x, s, xt32);
    pack_wqkv_kernel<<<dim3((MQKV * CC / 2) / 256), dim3(256), 0, stream>>>(wqkv, g, wg32);
    pack_wout_kernel<<<dim3((CC * HID / 2) / 256), dim3(256), 0, stream>>>(wout, wo32);

    qkv_kernel<<<dim3(NN / 64, MQKV / 128, BB), dim3(256), 0, stream>>>(
        xt32, wg32, s, qbuf, kbuf, vpbuf);
    attn_kernel<<<dim3(NN / 128, HEADS, BB), dim3(256), 0, stream>>>(
        qbuf, kbuf, vpbuf, obuf);
    out_kernel<<<dim3(NN / 64, CC / 128, BB), dim3(256), 0, stream>>>(
        wo32, bout, (const unsigned int*)obuf, y);
}
